// CrossAttentionBlock_22677427323236
// MI455X (gfx1250) — matrix operand order repaired and checked by reading, not compiled, not run on hardware
//
#include <hip/hip_runtime.h>
#include <hip/hip_bf16.h>

typedef __attribute__((ext_vector_type(16))) _Float16 v16h;
typedef __attribute__((ext_vector_type(8)))  _Float16 v8h;
typedef __attribute__((ext_vector_type(4)))  _Float16 v4h;
typedef __attribute__((ext_vector_type(8)))  float    v8f;

#define B_   8
#define C_   256
#define H_   64
#define W_   64
#define NH_  8
#define HD_  32
#define PH_  32
#define PW_  32
#define SQ_  1024          // pooled spatial = 32*32
#define VT_STRIDE 1032     // padded V-transpose row (bank-conflict mitigation)

static __device__ __forceinline__ v8f wmma_f16(v16h a, v16h b, v8f c) {
  // D = A(16x32 f16) * B(32x16 f16) + C(16x16 f32)
  return __builtin_amdgcn_wmma_f32_16x16x32_f16(false, a, false, b, (short)0, c,
                                                false, false);
}

// A operand: row-major 16x32; lane holds row m=lane%16, K segments {8h..8h+7, 16+8h..16+8h+7}
static __device__ __forceinline__ v16h load_a_f16(const _Float16* row, int half) {
  union { v16h v; v8h q[2]; } u;
  u.q[0] = *(const v8h*)(row + 8 * half);
  u.q[1] = *(const v8h*)(row + 16 + 8 * half);
  return u.v;
}
// B operand: 32x16; lane holds col n=lane%16, K segments {8h..8h+7, 16+8h..16+8h+7}
static __device__ __forceinline__ v16h load_b_f16(const _Float16* col, int half) {
  union { v16h v; v8h q[2]; } u;
  u.q[0] = *(const v8h*)(col + 8 * half);
  u.q[1] = *(const v8h*)(col + 16 + 8 * half);
  return u.v;
}

// ---------------- Kernel 0: convert all four weight matrices fp32 -> f16 --------------
__global__ void wcvt_kernel(const float* __restrict__ Wq, const float* __restrict__ Wk,
                            const float* __restrict__ Wv, const float* __restrict__ Wo,
                            _Float16* __restrict__ Whq, _Float16* __restrict__ Whk,
                            _Float16* __restrict__ Whv, _Float16* __restrict__ Who) {
  int w = blockIdx.y;
  const float* src = (w == 0) ? Wq : (w == 1) ? Wk : (w == 2) ? Wv : Wo;
  _Float16*    dst = (w == 0) ? Whq : (w == 1) ? Whk : (w == 2) ? Whv : Who;
  int i = (blockIdx.x * 256 + threadIdx.x) * 4;   // 64 blocks * 256 thr * 4 = 65536
  float4 v = *(const float4*)(src + i);
  v4h h; h.x = (_Float16)v.x; h.y = (_Float16)v.y; h.z = (_Float16)v.z; h.w = (_Float16)v.w;
  *(v4h*)(dst + i) = h;
}

// ---------------- Kernel 1: 2x2 avgpool + fp32->f16, layout [b][s][c] ----------------
__global__ void pool_half_kernel(const float* __restrict__ q, const float* __restrict__ kv,
                                 _Float16* __restrict__ qp, _Float16* __restrict__ kp) {
  int x = threadIdx.x & 31;
  int y = (blockIdx.x << 3) + (threadIdx.x >> 5);
  int c = blockIdx.y;
  int z = blockIdx.z;
  const float* src = (z < B_) ? q : kv;
  _Float16*    dst = (z < B_) ? qp : kp;
  int b = (z < B_) ? z : z - B_;
  const float* p = src + (((size_t)b * C_ + c) * H_ + 2 * y) * W_ + 2 * x;
  float s = 0.25f * (p[0] + p[1] + p[W_] + p[W_ + 1]);
  dst[((size_t)b * SQ_ + y * PW_ + x) * C_ + c] = (_Float16)s;
}

// ---------------- Kernel 2: QKV projections (GEMM 1024x256x256 per batch) -------------
// out[s, co] = sum_c in[s,c] * W[co,c] ; stored head-major [b][h][s][d] f16
__global__ void proj_kernel(const _Float16* __restrict__ qp, const _Float16* __restrict__ kp,
                            const _Float16* __restrict__ Whq, const _Float16* __restrict__ Whk,
                            const _Float16* __restrict__ Whv,
                            _Float16* __restrict__ Qh, _Float16* __restrict__ Kh,
                            _Float16* __restrict__ Vh) {
  int tid = threadIdx.x, lane = tid & 31, wv = tid >> 5;
  int n = lane & 15, half = lane >> 4;
  int mt = blockIdx.x;
  int b = mt >> 6, s0 = (mt & 63) << 4;
  int which = blockIdx.y;
  const _Float16* src = (which == 0) ? qp : kp;
  const _Float16* Wp  = (which == 0) ? Whq : (which == 1) ? Whk : Whv;
  _Float16*       dst = (which == 0) ? Qh : (which == 1) ? Kh : Vh;
  int nbase = wv << 5;  // each wave: 32 output channels = one head
  v8f c0 = {}, c1 = {};
  const _Float16* arow  = src + ((size_t)b * SQ_ + s0 + n) * C_;
  const _Float16* wrow0 = Wp + (size_t)(nbase + n) * C_;
  const _Float16* wrow1 = Wp + (size_t)(nbase + 16 + n) * C_;
#pragma unroll
  for (int kc = 0; kc < C_; kc += 32) {
    v16h a  = load_a_f16(arow + kc, half);
    v16h b0 = load_b_f16(wrow0 + kc, half);
    v16h b1 = load_b_f16(wrow1 + kc, half);
    c0 = wmma_f16(a, b0, c0);
    c1 = wmma_f16(a, b1, c1);
  }
  int head = wv;
#pragma unroll
  for (int r = 0; r < 8; ++r) {
    int srow = s0 + r + 8 * half;
    size_t base = (((size_t)b * NH_ + head) * SQ_ + srow) * HD_;
    dst[base + n]      = (_Float16)c0[r];
    dst[base + 16 + n] = (_Float16)c1[r];
  }
}

// ---------------- Kernel 3: flash attention, K/V resident in LDS ----------------------
// grid.x = b*h (64), grid.y = query blocks of 128; each wave owns 16 query rows.
__global__ __launch_bounds__(256, 1) void attn_kernel(const _Float16* __restrict__ Qh,
                                                      const _Float16* __restrict__ Kh,
                                                      const _Float16* __restrict__ Vh,
                                                      _Float16* __restrict__ O) {
  __shared__ _Float16 sK[SQ_ * HD_];          // 64 KB, row-major [key][d]
  __shared__ _Float16 sVt[HD_ * VT_STRIDE];   // 66 KB, transposed [d][key] (padded)
  __shared__ _Float16 sP[8][16 * 32];         // 8 KB, per-wave P tile staging
  int tid = threadIdx.x, lane = tid & 31, wv = tid >> 5;
  int n = lane & 15, half = lane >> 4;
  int bh = blockIdx.x;
  int q0 = (blockIdx.y * 8 + wv) * 16;
  const _Float16* qptr = Qh + ((size_t)bh * SQ_ + q0 + n) * HD_;
  __builtin_prefetch(qptr, 0, 1);  // global_prefetch_b8: Q tile in flight during staging

  const _Float16* Kb = Kh + (size_t)bh * SQ_ * HD_;
  const _Float16* Vb = Vh + (size_t)bh * SQ_ * HD_;
  for (int i = tid; i < SQ_ * HD_ / 8; i += 256)
    ((v8h*)sK)[i] = ((const v8h*)Kb)[i];
  for (int i = tid; i < SQ_ * HD_; i += 256) {
    int key = i >> 5, d = i & 31;
    sVt[(size_t)d * VT_STRIDE + key] = Vb[i];
  }
  __syncthreads();

  v16h qa = load_a_f16(qptr, half);
  const float scale = 0.17677669529663687f;  // 32^-0.5

  v8f acc0 = {}, acc1 = {};
  float mrow[8], lrow[8];
#pragma unroll
  for (int r = 0; r < 8; ++r) { mrow[r] = -1e30f; lrow[r] = 0.0f; }
  _Float16* pbuf = &sP[wv][0];

  for (int kc = 0; kc < SQ_; kc += 32) {
    v16h bk0 = load_b_f16(sK + (size_t)(kc + n) * HD_, half);
    v16h bk1 = load_b_f16(sK + (size_t)(kc + 16 + n) * HD_, half);
    v8f z = {};
    v8f s0 = wmma_f16(qa, bk0, z);
    v8f s1 = wmma_f16(qa, bk1, z);
    float t[8];
#pragma unroll
    for (int r = 0; r < 8; ++r) {
      s0[r] *= scale; s1[r] *= scale;
      t[r] = fmaxf(s0[r], s1[r]);
    }
    // row-max allreduce across the 16 lanes of each C-tile half-group
#pragma unroll
    for (int off = 1; off < 16; off <<= 1)
#pragma unroll
      for (int r = 0; r < 8; ++r) t[r] = fmaxf(t[r], __shfl_xor(t[r], off, 32));
#pragma unroll
    for (int r = 0; r < 8; ++r) {
      float nm = fmaxf(mrow[r], t[r]);
      float corr = __expf(mrow[r] - nm);
      mrow[r] = nm;
      lrow[r] *= corr; acc0[r] *= corr; acc1[r] *= corr;
      float p0 = __expf(s0[r] - nm);
      float p1 = __expf(s1[r] - nm);
      lrow[r] += p0 + p1;
      int mr = (r + 8 * half) * 32;   // C-layout row -> P-tile row
      pbuf[mr + n]      = (_Float16)p0;
      pbuf[mr + 16 + n] = (_Float16)p1;
    }
    // reload P as an A operand (same-wave DS ops are in-order)
    v16h pa  = load_a_f16(pbuf + n * 32, half);
    v16h bv0 = load_b_f16(sVt + (size_t)n * VT_STRIDE + kc, half);
    v16h bv1 = load_b_f16(sVt + (size_t)(16 + n) * VT_STRIDE + kc, half);
    acc0 = wmma_f16(pa, bv0, acc0);
    acc1 = wmma_f16(pa, bv1, acc1);
  }
  // row-sum allreduce, normalize, store O as [b][s][c] (c = h*32+d) for the Wo GEMM
#pragma unroll
  for (int off = 1; off < 16; off <<= 1)
#pragma unroll
    for (int r = 0; r < 8; ++r) lrow[r] += __shfl_xor(lrow[r], off, 32);
  int b = bh >> 3, h = bh & 7;
#pragma unroll
  for (int r = 0; r < 8; ++r) {
    float rinv = 1.0f / lrow[r];
    int srow = q0 + r + 8 * half;
    size_t base = ((size_t)b * SQ_ + srow) * C_ + (h << 5);
    O[base + n]      = (_Float16)(acc0[r] * rinv);
    O[base + 16 + n] = (_Float16)(acc1[r] * rinv);
  }
}

// ---------------- Kernel 4: Wo GEMM at pooled resolution (upsample commutes) ----------
__global__ void ogemm_kernel(const _Float16* __restrict__ O, const _Float16* __restrict__ Who,
                             float* __restrict__ Y) {
  int tid = threadIdx.x, lane = tid & 31, wv = tid >> 5;
  int n = lane & 15, half = lane >> 4;
  int mt = blockIdx.x;
  int b = mt >> 6, s0 = (mt & 63) << 4;
  int nbase = wv << 5;
  v8f c0 = {}, c1 = {};
  const _Float16* arow  = O + ((size_t)b * SQ_ + s0 + n) * C_;
  const _Float16* wrow0 = Who + (size_t)(nbase + n) * C_;
  const _Float16* wrow1 = Who + (size_t)(nbase + 16 + n) * C_;
#pragma unroll
  for (int kc = 0; kc < C_; kc += 32) {
    v16h a  = load_a_f16(arow + kc, half);
    v16h b0 = load_b_f16(wrow0 + kc, half);
    v16h b1 = load_b_f16(wrow1 + kc, half);
    c0 = wmma_f16(a, b0, c0);
    c1 = wmma_f16(a, b1, c1);
  }
  // store fp32 planes [b][co][s]; per-lane rows are contiguous in s -> 16B stores
  int sbase = s0 + 8 * half;
  size_t o0 = ((size_t)b * C_ + nbase + n) * SQ_ + sbase;
  size_t o1 = ((size_t)b * C_ + nbase + 16 + n) * SQ_ + sbase;
  *(float4*)(Y + o0)     = make_float4(c0[0], c0[1], c0[2], c0[3]);
  *(float4*)(Y + o0 + 4) = make_float4(c0[4], c0[5], c0[6], c0[7]);
  *(float4*)(Y + o1)     = make_float4(c1[0], c1[1], c1[2], c1[3]);
  *(float4*)(Y + o1 + 4) = make_float4(c1[4], c1[5], c1[6], c1[7]);
}

// ---------------- Kernel 5: bilinear 2x upsample + residual + BN (fused) --------------
__global__ void up_bn_kernel(const float* __restrict__ Y, const float* __restrict__ query,
                             const float* __restrict__ gamma, const float* __restrict__ beta,
                             const float* __restrict__ mean, const float* __restrict__ var,
                             float* __restrict__ out) {
  int X  = threadIdx.x & 63;
  int Yo = (blockIdx.x << 2) + (threadIdx.x >> 6);
  int c = blockIdx.y, b = blockIdx.z;
  int jy = Yo >> 1, jx = X >> 1;
  int ya, yb, xa, xb; float wya, wyb, wxa, wxb;
  if (Yo & 1) { ya = jy; yb = (jy + 1 < PH_) ? jy + 1 : PH_ - 1; wya = 0.75f; wyb = 0.25f; }
  else        { ya = (jy > 0) ? jy - 1 : 0; yb = jy;             wya = 0.25f; wyb = 0.75f; }
  if (X & 1)  { xa = jx; xb = (jx + 1 < PW_) ? jx + 1 : PW_ - 1; wxa = 0.75f; wxb = 0.25f; }
  else        { xa = (jx > 0) ? jx - 1 : 0; xb = jx;             wxa = 0.25f; wxb = 0.75f; }
  const float* Yp = Y + ((size_t)b * C_ + c) * SQ_;
  float up = wya * (wxa * Yp[ya * PW_ + xa] + wxb * Yp[ya * PW_ + xb])
           + wyb * (wxa * Yp[yb * PW_ + xa] + wxb * Yp[yb * PW_ + xb]);
  size_t idx = (((size_t)b * C_ + c) * H_ + Yo) * W_ + X;
  float g  = gamma[c] * rsqrtf(var[c] + 1e-5f);
  float bb = beta[c] - mean[c] * g;
  out[idx] = (up + query[idx]) * g + bb;
}

extern "C" void kernel_launch(void* const* d_in, const int* in_sizes, int n_in,
                              void* d_out, int out_size, void* d_ws, size_t ws_size,
                              hipStream_t stream) {
  (void)in_sizes; (void)n_in; (void)out_size; (void)ws_size;
  const float* query = (const float*)d_in[0];
  const float* kv    = (const float*)d_in[1];
  const float* Wq    = (const float*)d_in[2];
  const float* Wk    = (const float*)d_in[3];
  const float* Wv    = (const float*)d_in[4];
  const float* Wo    = (const float*)d_in[5];
  const float* gamma = (const float*)d_in[6];
  const float* beta  = (const float*)d_in[7];
  const float* mean  = (const float*)d_in[8];
  const float* var   = (const float*)d_in[9];
  // d_in[10] = num_heads (fixed 8 per setup_inputs)
  float* out = (float*)d_out;

  char* ws = (char*)d_ws;                        // 32.5 MB total, all fully overwritten
  _Float16* qp  = (_Float16*)(ws);               //  4 MB [b][s][c] f16
  _Float16* kp  = (_Float16*)(ws + (4u  << 20)); //  4 MB
  _Float16* Qh  = (_Float16*)(ws + (8u  << 20)); //  4 MB [b][h][s][d] f16
  _Float16* Kh  = (_Float16*)(ws + (12u << 20)); //  4 MB
  _Float16* Vh  = (_Float16*)(ws + (16u << 20)); //  4 MB
  _Float16* O   = (_Float16*)(ws + (20u << 20)); //  4 MB [b][s][c] f16
  float*    Y   = (float*)   (ws + (24u << 20)); //  8 MB [b][c][s] f32
  _Float16* Whq = (_Float16*)(ws + (32u << 20));             // 128 KB each, f16 weights
  _Float16* Whk = (_Float16*)(ws + (32u << 20) + (1u << 17));
  _Float16* Whv = (_Float16*)(ws + (32u << 20) + (2u << 17));
  _Float16* Who = (_Float16*)(ws + (32u << 20) + (3u << 17));

  wcvt_kernel<<<dim3(64, 4), 256, 0, stream>>>(Wq, Wk, Wv, Wo, Whq, Whk, Whv, Who);
  pool_half_kernel<<<dim3(4, 256, 16), 256, 0, stream>>>(query, kv, qp, kp);
  proj_kernel<<<dim3(512, 3), 256, 0, stream>>>(qp, kp, Whq, Whk, Whv, Qh, Kh, Vh);
  attn_kernel<<<dim3(64, 8), 256, 0, stream>>>(Qh, Kh, Vh, O);
  ogemm_kernel<<<dim3(512), 256, 0, stream>>>(O, Who, Y);
  up_bn_kernel<<<dim3(16, 256, 8), 256, 0, stream>>>(Y, query, gamma, beta, mean, var, out);
}
